// MultiHeadAttention_7438883356835
// MI455X (gfx1250) — matrix operand order repaired and checked by reading, not compiled, not run on hardware
//
#include <hip/hip_runtime.h>
#include <hip/hip_bf16.h>

// MHA: B=2, N=2048, C=1024, H=16, D=64.
// f16 WMMA (V_WMMA_F32_16X16X32_F16) with f32 accumulation.
//   1) cast x -> f16
//   2) cast+transpose Wq/Wk/Wv/Wo -> f16 Wt[out][in]
//   3) fused QKV GEMM (+bias) -> [B,H,N,D] f16   (async-LDS double-buffered)
//   4) flash attention (online softmax) -> [B,N,C] f16
//   5) output projection GEMM (+bias) -> f32 d_out
// Workspace: 48 MiB f16 scratch.

typedef __attribute__((ext_vector_type(16))) _Float16 v16h;
typedef __attribute__((ext_vector_type(8)))  _Float16 v8h;
typedef __attribute__((ext_vector_type(8)))  float    v8f;

#define LDSS 72   // LDS row stride in halves (144B: 16B-aligned, breaks bank conflicts)

#define NB   2
#define SEQ  2048
#define CDIM 1024
#define NH   16
#define HD   64
#define MROW 4096            // NB*SEQ
#define XH_ELEMS   4194304   // 4096*1024
#define WT_ELEMS   1048576   // 1024*1024
#define QKV_ELEMS  4194304   // 2*16*2048*64 per tensor

// ---------------------------------------------------------------------------
// CDNA5 async LDS copy (ASYNCcnt-tracked, no VGPR round trip).
// LDS operand is the low 32 bits of the generic pointer.
// ---------------------------------------------------------------------------
__device__ __forceinline__ void async_copy16(void* ldsPtr, const void* gPtr) {
  unsigned ldsOff = (unsigned)(unsigned long long)ldsPtr;
  asm volatile("global_load_async_to_lds_b128 %0, %1, off"
               :: "v"(ldsOff), "v"(gPtr) : "memory");
}
__device__ __forceinline__ void wait_async0() {
  asm volatile("s_wait_asynccnt 0x0" ::: "memory");
}

// ---------------------------------------------------------------------------
// WMMA helpers
// ---------------------------------------------------------------------------
__device__ __forceinline__ v8f wmma_f16(v16h a, v16h b, v8f c) {
  return __builtin_amdgcn_wmma_f32_16x16x32_f16(
      /*neg_a=*/false, a, /*neg_b=*/false, b,
      /*c_mod=*/(short)0, c, /*reuse_a=*/false, /*reuse_b=*/false);
}

// A-matrix fragment (16x32 f16), ISA layout:
//   lanes 0-15:  row M=lane,    K = {0..7, 16..23}
//   lanes 16-31: row M=lane-16, K = {8..15, 24..31}
__device__ __forceinline__ v16h frag_a(const _Float16* base, int lane, int k0) {
  const _Float16* p = base + (lane & 15) * LDSS + k0 + ((lane >> 4) << 3);
  v8h lo = *(const v8h*)p;
  v8h hi = *(const v8h*)(p + 16);
  return __builtin_shufflevector(lo, hi, 0,1,2,3,4,5,6,7,8,9,10,11,12,13,14,15);
}

// B-matrix fragment (32x16 f16) from a [col][k] LDS tile:
//   lanes 0-15: col N=lane, K={0..7, 16..23}; lanes 16-31: col N=lane-16, K={8..15, 24..31}
__device__ __forceinline__ v16h frag_b(const _Float16* base, int lane, int k0) {
  const _Float16* p = base + (lane & 15) * LDSS + k0 + ((lane >> 4) << 3);
  v8h lo = *(const v8h*)p;
  v8h hi = *(const v8h*)(p + 16);
  return __builtin_shufflevector(lo, hi, 0,1,2,3,4,5,6,7,8,9,10,11,12,13,14,15);
}

// ---------------------------------------------------------------------------
// 1) cast x -> f16
// ---------------------------------------------------------------------------
__global__ void mha_cast_x(const float* __restrict__ x, _Float16* __restrict__ xh) {
  int i = blockIdx.x * blockDim.x + threadIdx.x;
  if (i < XH_ELEMS) xh[i] = (_Float16)x[i];
}

// ---------------------------------------------------------------------------
// 2) cast + transpose weights: Wt[z][n][k] = (f16) W_z[k][n]
// ---------------------------------------------------------------------------
__global__ void mha_transpose_w(const float* __restrict__ Wq, const float* __restrict__ Wk,
                                const float* __restrict__ Wv, const float* __restrict__ Wo,
                                _Float16* __restrict__ Wt) {
  int z = blockIdx.z;
  const float* W = (z == 0) ? Wq : (z == 1) ? Wk : (z == 2) ? Wv : Wo;
  int i = blockIdx.x * blockDim.x + threadIdx.x;  // i = k*1024 + n
  if (i < WT_ELEMS) {
    int k = i >> 10, n = i & 1023;
    Wt[(size_t)z * WT_ELEMS + (size_t)n * CDIM + k] = (_Float16)W[i];
  }
}

// ---------------------------------------------------------------------------
// 3)/5) tiled GEMM: C[4096,1024] = A[4096,1024] x W + bias
//   block tile 128x64, 8 waves each 32x32 (2x2 WMMA tiles), K-tile 64,
//   double-buffered LDS filled by async-LDS copies.
// ---------------------------------------------------------------------------
__global__ void mha_gemm(const _Float16* __restrict__ A,
                         const _Float16* __restrict__ WtBase,
                         const float* __restrict__ b0, const float* __restrict__ b1,
                         const float* __restrict__ b2,
                         _Float16* __restrict__ outH, float* __restrict__ outF,
                         int finalProj) {
  __shared__ __align__(16) _Float16 As[2][128 * LDSS];
  __shared__ __align__(16) _Float16 Bs[2][64 * LDSS];

  const int tid  = threadIdx.x;
  const int lane = tid & 31;
  const int wave = tid >> 5;
  const int mode = blockIdx.z;
  const _Float16* Wt = WtBase + (size_t)mode * WT_ELEMS;
  const float* bias = (mode == 0) ? b0 : (mode == 1) ? b1 : b2;

  const int mBase = blockIdx.y * 128;
  const int nBase = blockIdx.x * 64;
  const int wr = (wave >> 1) * 32;
  const int wc = (wave & 1) * 32;

  v8f acc[2][2] = {};

  // async-stage one 128x64 A tile + 64x64 B tile into buffer `buf`
  auto stage = [&](int buf, int k0) {
#pragma unroll
    for (int i = 0; i < 4; ++i) {                 // A: 1024 chunks / 256 thr
      int ch = tid + i * 256;
      int r = ch >> 3, kc = (ch & 7) << 3;
      async_copy16(&As[buf][r * LDSS + kc],
                   A + (size_t)(mBase + r) * CDIM + k0 + kc);
    }
#pragma unroll
    for (int i = 0; i < 2; ++i) {                 // B: 512 chunks / 256 thr
      int ch = tid + i * 256;
      int c = ch >> 3, kc = (ch & 7) << 3;
      async_copy16(&Bs[buf][c * LDSS + kc],
                   Wt + (size_t)(nBase + c) * CDIM + k0 + kc);
    }
  };

  stage(0, 0);
  wait_async0();
  __syncthreads();

  for (int kt = 0; kt < CDIM / 64; ++kt) {
    const int cur = kt & 1;
    if (kt + 1 < CDIM / 64) stage(cur ^ 1, (kt + 1) * 64);
    if (kt + 2 < CDIM / 64) {  // pull tile kt+2 toward L2
      __builtin_prefetch(A + (size_t)(mBase + (tid >> 1)) * CDIM + (kt + 2) * 64, 0, 1);
      if (tid < 128)
        __builtin_prefetch(Wt + (size_t)(nBase + (tid >> 1)) * CDIM + (kt + 2) * 64, 0, 1);
    }

    const _Float16* Ab = As[cur];
    const _Float16* Bb = Bs[cur];
#pragma unroll
    for (int kk = 0; kk < 64; kk += 32) {
      v16h a0  = frag_a(Ab + (wr)      * LDSS, lane, kk);
      v16h a1  = frag_a(Ab + (wr + 16) * LDSS, lane, kk);
      v16h bb0 = frag_b(Bb + (wc)      * LDSS, lane, kk);
      v16h bb1 = frag_b(Bb + (wc + 16) * LDSS, lane, kk);
      acc[0][0] = wmma_f16(a0, bb0, acc[0][0]);
      acc[0][1] = wmma_f16(a0, bb1, acc[0][1]);
      acc[1][0] = wmma_f16(a1, bb0, acc[1][0]);
      acc[1][1] = wmma_f16(a1, bb1, acc[1][1]);
    }

    wait_async0();     // next tile's async copies complete
    __syncthreads();   // all waves done reading `cur` before it is re-staged
  }

  // epilogue. C/D layout: vgpr j, lanes 0-15 -> row j, lanes 16-31 -> row j+8
  const int half = lane >> 4, cl = lane & 15;
#pragma unroll
  for (int it = 0; it < 2; ++it)
#pragma unroll
    for (int jt = 0; jt < 2; ++jt)
#pragma unroll
      for (int j = 0; j < 8; ++j) {
        int r = mBase + wr + it * 16 + j + 8 * half;
        int c = nBase + wc + jt * 16 + cl;
        float val = acc[it][jt][j] + bias[c];
        if (finalProj) {
          outF[(size_t)r * CDIM + c] = val;
        } else {
          int hh = c >> 6, d = c & 63;
          int bb = r >> 11, n = r & 2047;
          outH[(size_t)mode * QKV_ELEMS +
               (((size_t)(bb * NH + hh) * SEQ + n) << 6) + d] = (_Float16)val;
        }
      }
}

// ---------------------------------------------------------------------------
// 4) flash attention. grid = (SEQ/128, B*H), 256 threads (8 waves x 16 rows).
// ---------------------------------------------------------------------------
__global__ void mha_flash_attn(const _Float16* __restrict__ qkv,
                               _Float16* __restrict__ attn) {
  __shared__ __align__(16) _Float16 Qs[128 * LDSS];
  __shared__ __align__(16) _Float16 Ks[64 * LDSS];   // [n][d] == Bt for Q*K^T
  __shared__ __align__(16) _Float16 Vt[64 * LDSS];   // [d][n] == Bt for P*V
  __shared__ __align__(16) _Float16 Ps[128 * LDSS];  // per-wave 16-row regions

  const int tid  = threadIdx.x;
  const int lane = tid & 31;
  const int wave = tid >> 5;
  const int by = blockIdx.y;            // b*NH + h
  const int b = by >> 4, h = by & 15;
  const int rowBase = blockIdx.x * 128;

  const _Float16* Qg = qkv + (size_t)by * SEQ * HD;
  const _Float16* Kg = qkv + (size_t)QKV_ELEMS + (size_t)by * SEQ * HD;
  const _Float16* Vg = qkv + (size_t)2 * QKV_ELEMS + (size_t)by * SEQ * HD;

  // stage Q, scaled by 1/sqrt(64) (exact in f16); needs regs, so synchronous
#pragma unroll
  for (int i = 0; i < 4; ++i) {
    int ch = tid + i * 256;
    int r = ch >> 3, dc = (ch & 7) << 3;
    v8h q = *(const v8h*)(Qg + (size_t)(rowBase + r) * HD + dc);
#pragma unroll
    for (int e = 0; e < 8; ++e) q[e] = q[e] * (_Float16)0.125f;
    *(v8h*)(Qs + r * LDSS + dc) = q;
  }

  v8f acc_o[4] = {};
  float mrow[8], lrow[8];
#pragma unroll
  for (int j = 0; j < 8; ++j) { mrow[j] = -1e30f; lrow[j] = 0.0f; }

  const _Float16* Qw = Qs + wave * 16 * LDSS;
  _Float16*       Pw = Ps + wave * 16 * LDSS;
  const int half = lane >> 4, cl = lane & 15;

  for (int jb = 0; jb < SEQ / 64; ++jb) {
    __syncthreads();   // all waves done reading Ks/Vt from previous block
    // K block: async straight into LDS [n][d]
#pragma unroll
    for (int i = 0; i < 2; ++i) {
      int ch = tid + i * 256;
      int n = ch >> 3, dc = (ch & 7) << 3;
      async_copy16(Ks + n * LDSS + dc,
                   Kg + (size_t)(jb * 64 + n) * HD + dc);
    }
    // V block: through regs (transposed store [d][n])
#pragma unroll
    for (int i = 0; i < 2; ++i) {
      int ch = tid + i * 256;
      int n = ch >> 3, dc = (ch & 7) << 3;
      v8h vv = *(const v8h*)(Vg + (size_t)(jb * 64 + n) * HD + dc);
#pragma unroll
      for (int e = 0; e < 8; ++e) Vt[(dc + e) * LDSS + n] = vv[e];
    }
    if (jb + 1 < SEQ / 64) {  // prefetch next K/V block toward L2
      __builtin_prefetch(Kg + (size_t)(jb + 1) * 64 * HD + tid * 16, 0, 1);
      __builtin_prefetch(Vg + (size_t)(jb + 1) * 64 * HD + tid * 16, 0, 1);
    }
    wait_async0();
    __syncthreads();

    // S = (Q/sqrt(D)) K^T : 16x64 per wave
    v8f s[4] = {};
#pragma unroll
    for (int kk = 0; kk < HD; kk += 32) {
      v16h a = frag_a(Qw, lane, kk);
#pragma unroll
      for (int t = 0; t < 4; ++t) {
        v16h bf = frag_b(Ks + t * 16 * LDSS, lane, kk);
        s[t] = wmma_f16(a, bf, s[t]);
      }
    }

    // online softmax (shuffle masks 1..8 reduce the 16 cols in each lane half)
    float sc[8];
#pragma unroll
    for (int j = 0; j < 8; ++j) {
      float mx = fmaxf(fmaxf(s[0][j], s[1][j]), fmaxf(s[2][j], s[3][j]));
#pragma unroll
      for (int off = 1; off < 16; off <<= 1)
        mx = fmaxf(mx, __shfl_xor(mx, off, 32));
      float mnew = fmaxf(mrow[j], mx);
      sc[j] = __expf(mrow[j] - mnew);
      mrow[j] = mnew;
      float rs = 0.0f;
#pragma unroll
      for (int t = 0; t < 4; ++t) {
        float p = __expf(s[t][j] - mnew);
        s[t][j] = p;
        rs += p;
      }
#pragma unroll
      for (int off = 1; off < 16; off <<= 1)
        rs += __shfl_xor(rs, off, 32);
      lrow[j] = lrow[j] * sc[j] + rs;
    }
#pragma unroll
    for (int t = 0; t < 4; ++t)
#pragma unroll
      for (int j = 0; j < 8; ++j) acc_o[t][j] *= sc[j];

    // P: C-layout -> row-major LDS (per-wave region; DS in-order within wave)
#pragma unroll
    for (int t = 0; t < 4; ++t)
#pragma unroll
      for (int j = 0; j < 8; ++j)
        Pw[(j + 8 * half) * LDSS + t * 16 + cl] = (_Float16)s[t][j];

    // O += P V
#pragma unroll
    for (int kk = 0; kk < 64; kk += 32) {
      v16h a = frag_a(Pw, lane, kk);
#pragma unroll
      for (int t = 0; t < 4; ++t) {
        v16h bf = frag_b(Vt + t * 16 * LDSS, lane, kk);
        acc_o[t] = wmma_f16(a, bf, acc_o[t]);
      }
    }
  }

  // normalize and write [B,N,C] f16
#pragma unroll
  for (int t = 0; t < 4; ++t)
#pragma unroll
    for (int j = 0; j < 8; ++j) {
      int n = rowBase + wave * 16 + j + 8 * half;
      int d = t * 16 + cl;
      float val = acc_o[t][j] / lrow[j];
      attn[((size_t)(b * SEQ + n)) * CDIM + h * HD + d] = (_Float16)val;
    }
}

// ---------------------------------------------------------------------------
// launch
// ---------------------------------------------------------------------------
extern "C" void kernel_launch(void* const* d_in, const int* in_sizes, int n_in,
                              void* d_out, int out_size, void* d_ws, size_t ws_size,
                              hipStream_t stream) {
  const float* x  = (const float*)d_in[0];
  const float* Wq = (const float*)d_in[1];
  const float* bq = (const float*)d_in[2];
  const float* Wk = (const float*)d_in[3];
  const float* bk = (const float*)d_in[4];
  const float* Wv = (const float*)d_in[5];
  const float* bv = (const float*)d_in[6];
  const float* Wo = (const float*)d_in[7];
  const float* bo = (const float*)d_in[8];
  float* out = (float*)d_out;

  _Float16* ws   = (_Float16*)d_ws;
  _Float16* xh   = ws;                           // 4,194,304
  _Float16* wt   = xh + XH_ELEMS;                // 4 x 1,048,576
  _Float16* qkvh = wt + (size_t)4 * WT_ELEMS;    // 3 x 4,194,304
  _Float16* atth = qkvh + (size_t)3 * QKV_ELEMS; // 4,194,304

  mha_cast_x<<<XH_ELEMS / 256, 256, 0, stream>>>(x, xh);
  mha_transpose_w<<<dim3(WT_ELEMS / 256, 1, 4), 256, 0, stream>>>(Wq, Wk, Wv, Wo, wt);

  mha_gemm<<<dim3(CDIM / 64, MROW / 128, 3), 256, 0, stream>>>(
      xh, wt, bq, bk, bv, qkvh, nullptr, 0);

  mha_flash_attn<<<dim3(SEQ / 128, NB * NH), 256, 0, stream>>>(qkvh, atth);

  mha_gemm<<<dim3(CDIM / 64, MROW / 128, 1), 256, 0, stream>>>(
      atth, wt + (size_t)3 * WT_ELEMS, bo, bo, bo, nullptr, out, 1);
}
